// GroupedBiSpectralScan_46437186404684
// MI455X (gfx1250) — matrix operand order repaired and checked by reading, not compiled, not run on hardware
//
#include <hip/hip_runtime.h>
#include <hip/hip_bf16.h>

// ---------------- problem constants ----------------
#define NB     512
#define BANDS  274
#define DD     256
#define GG     8
#define KK     7
#define LL     4
#define HH     512          // 2*DD
#define TWOH   1024
#define EPSLN  1e-5f

#define WREG    32          // bands per workgroup region
#define NROWS_A 40          // u rows in LDS; rows 38/39 are write-scratch (never read)
#define TILES_A 3
#define TILES_B 2

typedef __bf16 v16bf __attribute__((ext_vector_type(16)));
typedef __bf16 v8bf  __attribute__((ext_vector_type(8)));
typedef float  v8f   __attribute__((ext_vector_type(8)));
typedef float  v4f   __attribute__((ext_vector_type(4)));

__device__ __forceinline__ float fsig(float x) {
    return __builtin_amdgcn_rcpf(1.f + __expf(-x));   // v_exp + v_rcp
}

__device__ __forceinline__ v8f wmma_bf16(v16bf a, v16bf b, v8f c) {
    // D = A(16x32 bf16) * B(32x16 bf16) + C(16x16 f32)
    return __builtin_amdgcn_wmma_f32_16x16x32_bf16(false, a, false, b,
                                                   (short)0, c, false, false);
}

// B fragment: col n = lane&15; K chunks {kb..kb+7, kb+16..kb+23}, kb = (lane>>4)*8.
// weights stored row-major [out][in] -> two 8-bf16 chunks per lane (16B loads)
__device__ __forceinline__ v16bf load_bfrag(const __bf16* __restrict__ base,
                                            int rowstride, int kadd, int lane) {
    int n = lane & 15;
    int k = kadd + ((lane >> 4) << 3);
    const __bf16* p = base + (size_t)n * rowstride + k;
    v8bf lo = *(const v8bf*)(p);
    v8bf hi = *(const v8bf*)(p + 16);
    v16bf u;
#pragma unroll
    for (int t = 0; t < 8; ++t) { u[t] = lo[t]; u[8 + t] = hi[t]; }
    return u;
}

// A fragment from bf16 LDS tile: row = lane&15, K chunks {kb..kb+7, kb+16..kb+23}
__device__ __forceinline__ v16bf load_afrag(const __bf16* base,
                                            int rowstride, int kadd, int lane) {
    int row = lane & 15;
    int kb  = ((lane >> 4) << 3) + kadd;
    const __bf16* p = base + row * rowstride + kb;
    v8bf lo = *(const v8bf*)(p);        // 16B ds load
    v8bf hi = *(const v8bf*)(p + 16);   // 16B ds load
    v16bf u;
#pragma unroll
    for (int t = 0; t < 8; ++t) { u[t] = lo[t]; u[8 + t] = hi[t]; }
    return u;
}

// ---------------- weight prep: f32 -> bf16, symmetrize+transpose conv taps ----------------
#define N_INW  131072   // L*G*128*32
#define N_MIXW 131072   // L*8*64*64
#define N_OUTW 524288   // L*256*512
#define N_WSYM 14336    // L*7*512   (tap-major, channel contiguous)
#define N_CONV (N_INW + N_MIXW + N_OUTW + N_WSYM)

__global__ void prep_weights(const float* __restrict__ inw,  const float* __restrict__ mixw,
                             const float* __restrict__ outw, const float* __restrict__ dww,
                             __bf16* __restrict__ inw_bf,
                             __bf16* __restrict__ mixw_bf,
                             __bf16* __restrict__ outw_bf,
                             float* __restrict__ wsymT) {
    int i = blockIdx.x * blockDim.x + threadIdx.x;
    if (i < N_INW) {
        inw_bf[i] = (__bf16)inw[i];
    } else if (i < N_INW + N_MIXW) {
        int j = i - N_INW; mixw_bf[j] = (__bf16)mixw[j];
    } else if (i < N_INW + N_MIXW + N_OUTW) {
        int j = i - N_INW - N_MIXW; outw_bf[j] = (__bf16)outw[j];
    } else if (i < N_CONV) {
        int j  = i - N_INW - N_MIXW - N_OUTW;   // [l][t][ch]
        int ch = j % HH;
        int t  = (j / HH) % KK;
        int l  = j / (HH * KK);
        const float* d = dww + ((size_t)l * HH + ch) * KK;
        wsymT[j] = 0.5f * (d[t] + d[KK - 1 - t]);
    }
}

// ---------------- init: h = x*w_in + b_in + pos ----------------
__global__ void init_h(const float* __restrict__ x, const float* __restrict__ w_in,
                       const float* __restrict__ b_in, const float* __restrict__ pos,
                       float* __restrict__ h) {
    long long idx = (long long)blockIdx.x * blockDim.x + threadIdx.x;
    if (idx < (long long)NB * BANDS * DD) {
        int d = (int)(idx & (DD - 1));
        long long r = idx >> 8;
        int band = (int)(r % BANDS);
        int b    = (int)(r / BANDS);
        h[idx] = x[b * BANDS + band] * w_in[d] + b_in[d] + pos[band * DD + d];
    }
}

// ---------------- one full layer, fused, WMMA everywhere ----------------
__global__ __launch_bounds__(256)
void layer_kernel(const float* __restrict__ h_in, float* __restrict__ h_out,
                  const float* __restrict__ ln_g, const float* __restrict__ ln_b,
                  const float* __restrict__ inb,  const float* __restrict__ mixb,
                  const float* __restrict__ outb,
                  const __bf16* __restrict__ inw_bf,
                  const __bf16* __restrict__ mixw_bf,
                  const __bf16* __restrict__ outw_bf,
                  const float* __restrict__ wsymT, int l) {
    __shared__ __bf16 u_buf[NROWS_A][HH];   // 40 KB  (SiLU'd in-proj)
    __shared__ __bf16 um_tile[16][HH];      // 16 KB  (gated mix output)
    __shared__ float mu_s[NROWS_A];
    __shared__ float rs_s[NROWS_A];

    const int region = blockIdx.x;
    const int b      = blockIdx.y;
    const int s      = region * WREG;
    const int tid    = threadIdx.x;
    const int wave   = tid >> 5;      // wave32
    const int lane   = tid & 31;
    const bool edge  = (s == 0) || (s + NROWS_A - 3 > BANDS);   // regions 0 and 8

    // region-uniform valid rowA window: band = s-3+rowA must be in [0,BANDS),
    // and rowA < 38 (rows 38/39 are scratch). Stores outside go to row 39.
    int row_lo = 3 - s;              if (row_lo < 0) row_lo = 0;
    int row_hi = BANDS + 3 - s;      if (row_hi > NROWS_A - 2) row_hi = NROWS_A - 2;
    const unsigned row_span = (unsigned)(row_hi - row_lo);

    const float* lngL  = ln_g + l * DD;
    const float* lnbL  = ln_b + l * DD;
    const float* inbL  = inb  + l * TWOH;
    const float* mixbL = mixb + l * HH;
    const float* outbL = outb + l * DD;
    const __bf16* inwL  = inw_bf  + (size_t)l * GG * 128 * 32;
    const __bf16* mixwL = mixw_bf + (size_t)l * GG * 64 * 64;
    const __bf16* outwL = outw_bf + (size_t)l * 256 * 512;
    const float* wsymL = wsymT + (size_t)l * KK * HH;   // [t][ch]

    // ---- stage 0: pre-zero u_buf on edge regions (OOB conv taps must read 0) ----
    if (edge) {
        uint4* p = (uint4*)&u_buf[0][0];            // 2560 x 16B
        for (int i = tid; i < (int)(sizeof(u_buf) / 16); i += 256)
            p[i] = uint4{0u, 0u, 0u, 0u};
    }

    // ---- LayerNorm stats for bands [s-3, s+37) ----
    if (tid < NROWS_A) {
        int band = s - 3 + tid;
        float mu = 0.f, rs = 0.f;
        if (band >= 0 && band < BANDS) {
            const v4f* hp = (const v4f*)(h_in + ((size_t)b * BANDS + band) * DD);
            float sum = 0.f;
            for (int d = 0; d < DD / 4; ++d) {
                v4f v = hp[d];
                sum += v[0] + v[1] + v[2] + v[3];
            }
            mu = sum * (1.f / DD);
            float var = 0.f;
            for (int d = 0; d < DD / 4; ++d) {
                v4f v = hp[d];
#pragma unroll
                for (int t = 0; t < 4; ++t) { float dv = v[t] - mu; var += dv * dv; }
            }
            rs = rsqrtf(var * (1.f / DD) + EPSLN);
        }
        mu_s[tid] = mu; rs_s[tid] = rs;
    }
    __syncthreads();

    // A fragment of LN(h), dims [c0,c0+32). Branchless: band/stat clamped; rows that
    // would be invalid only feed results that are redirected/discarded later.
    auto make_y_frag = [&](int band0, int c0) -> v16bf {
        int row  = lane & 15;
        int kb   = (lane >> 4) << 3;
        int band = band0 + row;
        band = band < 0 ? 0 : (band >= BANDS ? BANDS - 1 : band);
        int ra = band - (s - 3);
        ra = ra < NROWS_A ? ra : NROWS_A - 1;
        float mu = mu_s[ra], rs = rs_s[ra];
        const float* hp = h_in + ((size_t)b * BANDS + band) * DD + c0 + kb;
        v16bf u;
#pragma unroll
        for (int t = 0; t < 8; ++t) {
            float y0 = (hp[t]      - mu) * rs * lngL[c0 + kb + t]      + lnbL[c0 + kb + t];
            float y1 = (hp[16 + t] - mu) * rs * lngL[c0 + kb + 16 + t] + lnbL[c0 + kb + 16 + t];
            u[t]     = (__bf16)y0;
            u[8 + t] = (__bf16)y1;
        }
        return u;
    };

    // ---- phase A: in-proj groups 0..3 -> SiLU -> u_buf (bands s-3 .. s+34) ----
    // wave w owns u channels [w*64, w*64+64)  (in-proj group g = w>>1)
    {
        const int g = wave >> 1;
        const __bf16* wg = inwL + (size_t)g * 128 * 32;
#pragma unroll 1
        for (int ta = 0; ta < TILES_A; ++ta) {
            v16bf a = make_y_frag(s - 3 + ta * 16, g * 32);
#pragma unroll
            for (int j = 0; j < 4; ++j) {
                int o0 = (wave & 1) * 64 + j * 16;
                v16bf bb = load_bfrag(wg + (size_t)o0 * 32, 32, 0, lane);
                v8f c = {};
                c = wmma_bf16(a, bb, c);
                int h_ch = wave * 64 + j * 16 + (lane & 15);
                float bias = inbL[h_ch];
                int mtop = (lane >> 4) << 3;
#pragma unroll
                for (int r = 0; r < 8; ++r) {
                    int rowA = ta * 16 + r + mtop;
                    float xv = c[r] + bias;
                    float uv = xv * fsig(xv);           // unconditional
                    // single unsigned range test, no branches:
                    int rowS = ((unsigned)(rowA - row_lo) < row_span)
                                   ? rowA : (NROWS_A - 1);
                    u_buf[rowS][h_ch] = (__bf16)uv;      // unconditional ds store
                }
            }
        }
    }
    __syncthreads();

    // ---- phase B: conv -> mix -> *sigmoid(gate) -> out-proj -> residual ----
#pragma unroll 1
    for (int tb = 0; tb < TILES_B; ++tb) {
        const int gm = wave;                   // mix group == wave
        const int chbase = gm * 64;

        // conv A fragment: tap-outer, 8-wide vector loads (LDS) and taps (global)
        auto conv_frag = [&](int kofs) -> v16bf {
            int row = lane & 15;
            int kb  = (lane >> 4) << 3;
            int r0  = tb * 16 + row;           // taps read rows r0..r0+6 (max 37)
            int ch0 = chbase + kofs + kb;
            int ch1 = ch0 + 16;
            float acc[16];
#pragma unroll
            for (int e = 0; e < 16; ++e) acc[e] = 0.f;
#pragma unroll
            for (int t = 0; t < KK; ++t) {
                v8bf a0 = *(const v8bf*)&u_buf[r0 + t][ch0];
                v8bf a1 = *(const v8bf*)&u_buf[r0 + t][ch1];
                v8f  w0 = *(const v8f*)(wsymL + (size_t)t * HH + ch0);
                v8f  w1 = *(const v8f*)(wsymL + (size_t)t * HH + ch1);
#pragma unroll
                for (int e = 0; e < 8; ++e) {
                    acc[e]     += w0[e] * (float)a0[e];
                    acc[8 + e] += w1[e] * (float)a1[e];
                }
            }
            v16bf u;
#pragma unroll
            for (int e = 0; e < 16; ++e) u[e] = (__bf16)acc[e];
            return u;
        };
        v16bf ac0 = conv_frag(0);
        v16bf ac1 = conv_frag(32);

        // gate: in-proj group gp = 4 + gm/2, columns (gm&1)*64 .. +64
        const int gp = 4 + (gm >> 1);
        v16bf ag = make_y_frag(s + tb * 16, gp * 32);
        const __bf16* wmix = mixwL + (size_t)gm * 64 * 64;
        const __bf16* wgte = inwL  + (size_t)gp * 128 * 32;

#pragma unroll
        for (int j = 0; j < 4; ++j) {
            v8f cm = {};
            cm = wmma_bf16(ac0, load_bfrag(wmix + (size_t)(j * 16) * 64, 64, 0,  lane), cm);
            cm = wmma_bf16(ac1, load_bfrag(wmix + (size_t)(j * 16) * 64, 64, 32, lane), cm);
            v8f cg = {};
            int og = (gm & 1) * 64 + j * 16;
            cg = wmma_bf16(ag, load_bfrag(wgte + (size_t)og * 32, 32, 0, lane), cg);

            int n = lane & 15;
            int h_ch = chbase + j * 16 + n;
            float mb = mixbL[h_ch];
            float gb = inbL[HH + h_ch];
            int mtop = (lane >> 4) << 3;
#pragma unroll
            for (int r = 0; r < 8; ++r) {
                float um = cm[r] + mb;
                float gt = cg[r] + gb;
                um_tile[r + mtop][h_ch] = (__bf16)(um * fsig(gt));
            }
        }
        __syncthreads();

        // out-proj: H=512 -> D=256; 16 N-tiles over 8 waves (2 each), 16 K-steps
#pragma unroll 1
        for (int jj = 0; jj < 2; ++jj) {
            int n0 = wave * 32 + jj * 16;
            v8f co = {};
#pragma unroll 4
            for (int ks = 0; ks < 16; ++ks) {
                v16bf a  = load_afrag(&um_tile[0][0], HH, ks * 32, lane);
                v16bf bb = load_bfrag(outwL + (size_t)n0 * 512, 512, ks * 32, lane);
                co = wmma_bf16(a, bb, co);
            }
            int d = n0 + (lane & 15);
            float ob = outbL[d];
            int mtop = (lane >> 4) << 3;
            if (s + WREG <= BANDS) {                 // uniform: interior regions
#pragma unroll
                for (int r = 0; r < 8; ++r) {
                    size_t off = ((size_t)b * BANDS + (s + tb * 16 + r + mtop)) * DD + d;
                    h_out[off] = h_in[off] + co[r] + ob;
                }
            } else {                                  // uniform: last region
#pragma unroll
                for (int r = 0; r < 8; ++r) {
                    int band = s + tb * 16 + r + mtop;
                    if (band < BANDS) {
                        size_t off = ((size_t)b * BANDS + band) * DD + d;
                        h_out[off] = h_in[off] + co[r] + ob;
                    }
                }
            }
        }
        __syncthreads();   // um_tile reused next tb
    }
}

// ---------------- final LayerNorm (in place on d_out) ----------------
__global__ __launch_bounds__(256)
void final_ln(float* __restrict__ h, const float* __restrict__ g,
              const float* __restrict__ bb) {
    __shared__ float red[256];
    int row = blockIdx.x;
    int t = threadIdx.x;
    size_t off = (size_t)row * DD + t;
    float v = h[off];
    red[t] = v; __syncthreads();
    for (int o = 128; o > 0; o >>= 1) { if (t < o) red[t] += red[t + o]; __syncthreads(); }
    float mu = red[0] * (1.f / DD);
    __syncthreads();
    float dv = v - mu;
    red[t] = dv * dv; __syncthreads();
    for (int o = 128; o > 0; o >>= 1) { if (t < o) red[t] += red[t + o]; __syncthreads(); }
    float rs = rsqrtf(red[0] * (1.f / DD) + EPSLN);
    h[off] = dv * rs * g[t] + bb[t];
}

// ---------------- host ----------------
extern "C" void kernel_launch(void* const* d_in, const int* in_sizes, int n_in,
                              void* d_out, int out_size, void* d_ws, size_t ws_size,
                              hipStream_t stream) {
    const float* x     = (const float*)d_in[0];
    const float* w_in  = (const float*)d_in[1];
    const float* b_in  = (const float*)d_in[2];
    const float* pos   = (const float*)d_in[3];
    const float* ln_g  = (const float*)d_in[4];
    const float* ln_b  = (const float*)d_in[5];
    const float* inw   = (const float*)d_in[6];
    const float* inb   = (const float*)d_in[7];
    const float* dww   = (const float*)d_in[8];
    const float* mixw  = (const float*)d_in[9];
    const float* mixb  = (const float*)d_in[10];
    const float* outw  = (const float*)d_in[11];
    const float* outb  = (const float*)d_in[12];
    const float* oln_g = (const float*)d_in[13];
    const float* oln_b = (const float*)d_in[14];

    float* out = (float*)d_out;

    // workspace layout: bf16 weights | wsymT f32 | ping-pong h buffer
    __bf16* inw_bf  = (__bf16*)d_ws;
    __bf16* mixw_bf = inw_bf + N_INW;
    __bf16* outw_bf = mixw_bf + N_MIXW;
    float* wsymT = (float*)((char*)d_ws + 2ull * (N_INW + N_MIXW + N_OUTW));
    float* h_alt = (float*)((char*)d_ws + 1703936ull);   // past weights, 64B aligned

    prep_weights<<<(N_CONV + 255) / 256, 256, 0, stream>>>(
        inw, mixw, outw, dww, inw_bf, mixw_bf, outw_bf, wsymT);

    const long long NTOT = (long long)NB * BANDS * DD;
    init_h<<<(unsigned)((NTOT + 255) / 256), 256, 0, stream>>>(x, w_in, b_in, pos, out);

    float* bufs[2] = { out, h_alt };
    const int nregions = (BANDS + WREG - 1) / WREG;   // 9
    for (int l = 0; l < LL; ++l) {
        layer_kernel<<<dim3(nregions, NB), 256, 0, stream>>>(
            bufs[l & 1], bufs[(l + 1) & 1],
            ln_g, ln_b, inb, mixb, outb,
            inw_bf, mixw_bf, outw_bf, wsymT, l);
    }
    // LL even -> final h is in bufs[0] == d_out
    final_ln<<<NB * BANDS, 256, 0, stream>>>(out, oln_g, oln_b);
}
